// GACN_22041772163119
// MI455X (gfx1250) — matrix operand order repaired and checked by reading, not compiled, not run on hardware
//
#include <hip/hip_runtime.h>

typedef __attribute__((ext_vector_type(16))) _Float16 v16h;
typedef __attribute__((ext_vector_type(8)))  _Float16 v8h;
typedef __attribute__((ext_vector_type(8)))  float    v8f;

#define NB 32   // batch
#define NC 64   // channels (C_IN == C_OUT)
#define NN 1024 // nodes
#define NT 16   // timesteps

// ---------------------------------------------------------------------------
// prep1: k[b,n,t] = sum_i x[b,i,n,t]*alpha[i];  q[b,n,s] = sum_t k[b,n,t]*W_att[t,s]
// ---------------------------------------------------------------------------
__global__ __launch_bounds__(256) void gacn_prep1(const float* __restrict__ x,
                                                  const float* __restrict__ alpha,
                                                  const float* __restrict__ Watt,
                                                  float* __restrict__ kbuf,
                                                  float* __restrict__ qbuf) {
    const int b = blockIdx.y, ng = blockIdx.x, tid = threadIdx.x;
    const int nn = tid >> 4, t = tid & 15;
    const int n  = ng * 16 + nn;
    __shared__ float s_alpha[NC];
    __shared__ float s_W[NT * NT];
    __shared__ float s_k[16][NT];
    if (tid < NC) s_alpha[tid] = alpha[tid];
    s_W[tid] = Watt[tid];
    __syncthreads();
    const float* xp = x + (size_t)b * NC * NN * NT + (size_t)n * NT + t;
    float acc = 0.f;
    #pragma unroll 4
    for (int i = 0; i < NC; ++i) acc += xp[(size_t)i * NN * NT] * s_alpha[i];
    kbuf[((size_t)b * NN + n) * NT + t] = acc;
    s_k[nn][t] = acc;
    __syncthreads();
    float qacc = 0.f;
    #pragma unroll
    for (int j = 0; j < NT; ++j) qacc += s_k[nn][j] * s_W[j * NT + t];
    qbuf[((size_t)b * NN + n) * NT + t] = qacc;
}

// ---------------------------------------------------------------------------
// prep2: yT[b][o][t][m] = sum_c W_out[o][c] * x[b][c][m][t]  (f16, t-major so
// the WMMA B-operand loads are contiguous per lane).
// ---------------------------------------------------------------------------
__global__ __launch_bounds__(256) void gacn_prep2(const float* __restrict__ x,
                                                  const float* __restrict__ Wout,
                                                  _Float16* __restrict__ yT) {
    const int b = blockIdx.y, mg = blockIdx.x, tid = threadIdx.x; // mg: 128 groups of 8 m
    __shared__ float s_x[NC * 8 * NT]; // [c][m(8)][t] = 32 KB
    const float* xb = x + (size_t)b * NC * NN * NT;
    for (int e = tid; e < NC * 8 * NT; e += 256) {
        int c = e >> 7, mt = e & 127;
        s_x[e] = xb[(size_t)c * NN * NT + (size_t)mg * 128 + mt];
    }
    __syncthreads();
    const int m = tid & 7, t = (tid >> 3) & 15, oh = tid >> 7; // oh wave-uniform
    for (int ob = 0; ob < 32; ob += 8) {
        float acc[8] = {0.f,0.f,0.f,0.f,0.f,0.f,0.f,0.f};
        for (int c = 0; c < NC; ++c) {
            float xv = s_x[c * 128 + m * 16 + t];
            #pragma unroll
            for (int j = 0; j < 8; ++j)
                acc[j] += Wout[(oh * 32 + ob + j) * NC + c] * xv;
        }
        #pragma unroll
        for (int j = 0; j < 8; ++j) {
            int o = oh * 32 + ob + j;
            yT[(((size_t)b * NC + o) * NT + t) * NN + (size_t)mg * 8 + m] = (_Float16)acc[j];
        }
    }
}

// ---------------------------------------------------------------------------
// att: one block per (b,n) row: fp32 scores + softmax, * adj, store f16.
// ---------------------------------------------------------------------------
__global__ __launch_bounds__(256) void gacn_att(const float* __restrict__ kbuf,
                                                const float* __restrict__ qbuf,
                                                const float* __restrict__ adj,
                                                _Float16* __restrict__ att) {
    const int b = blockIdx.y, n = blockIdx.x, tid = threadIdx.x;
    __shared__ float s_q[NT];
    __shared__ float s_s[NN];
    __shared__ float rbuf[256];
    if (tid < NT) s_q[tid] = qbuf[((size_t)b * NN + n) * NT + tid];
    __syncthreads();
    float lmax = -3.4e38f;
    for (int m = tid; m < NN; m += 256) {
        const float* kr = kbuf + ((size_t)b * NN + m) * NT;
        float s = 0.f;
        #pragma unroll
        for (int j = 0; j < NT; ++j) s += s_q[j] * kr[j];
        s_s[m] = s;
        lmax = fmaxf(lmax, s);
    }
    rbuf[tid] = lmax; __syncthreads();
    for (int w = 128; w > 0; w >>= 1) {
        if (tid < w) rbuf[tid] = fmaxf(rbuf[tid], rbuf[tid + w]);
        __syncthreads();
    }
    const float mx = rbuf[0]; __syncthreads();
    float lsum = 0.f;
    for (int m = tid; m < NN; m += 256) {
        float e = expf(s_s[m] - mx);
        s_s[m] = e;
        lsum += e;
    }
    rbuf[tid] = lsum; __syncthreads();
    for (int w = 128; w > 0; w >>= 1) {
        if (tid < w) rbuf[tid] += rbuf[tid + w];
        __syncthreads();
    }
    const float inv = 1.f / rbuf[0];
    for (int m = tid; m < NN; m += 256)
        att[((size_t)b * NN + n) * NN + m] =
            (_Float16)(s_s[m] * inv * adj[(size_t)n * NN + m]);
}

// ---------------------------------------------------------------------------
// Main WMMA GEMM: out[b,o,n,t] = sum_m att[b,n,m] * yT[b,o,t,m]
// grid (32, 64 n-tiles, 2), block 128 = 4 waves; each wave: one 16n x 16t tile
// for 8 consecutive o. A-operand is double-buffered in LDS via
// global_load_async_to_lds_b128 (ASYNCcnt), B-operand streams from L2.
// ---------------------------------------------------------------------------
#define A_ROWPAD 40   // halves per LDS row (80 B -> 20-word stride, conflict-free)
#define A_BUFSZ  (16 * A_ROWPAD * 2)  // bytes per A buffer (1280)

__global__ __launch_bounds__(128) void gacn_wmma(const _Float16* __restrict__ att,
                                                 const _Float16* __restrict__ yT,
                                                 float* __restrict__ out) {
    const int b    = blockIdx.x;
    const int n0   = blockIdx.y * 16;
    const int wave = threadIdx.x >> 5;
    const int lane = threadIdx.x & 31;
    const int o0   = (blockIdx.z * 4 + wave) * 8;     // 8 outputs per wave
    const int r    = lane & 15, hi = lane >> 4;

    __shared__ __align__(16) _Float16 s_a[4][2][16][A_ROWPAD]; // 4 waves x 2 bufs

    const _Float16* aBase = att + ((size_t)b * NN + n0) * NN;  // 16 rows, stride NN
    const _Float16* yBase = yT + (((size_t)b * NC + o0) * NT + r) * NN;
    const size_t oStride  = (size_t)NT * NN;

    // Per-lane async-copy task: 16 rows x 4 chunks of 16 B = 64 tasks, 2 instrs
    const int row0 = lane >> 2,        chk0 = lane & 3;         // tasks  0..31
    const int row1 = (32 + lane) >> 2, chk1 = (32 + lane) & 3;  // tasks 32..63
    const unsigned ldsWave = (unsigned)(uintptr_t)&s_a[wave][0][0][0];

    v8f acc[8];
    #pragma unroll
    for (int oi = 0; oi < 8; ++oi) acc[oi] = (v8f){};

    union H { v16h v; v8h h[2]; };

    // ---- prologue: async-copy A tile for m0 = 0 into buffer 0
    {
        const _Float16* g0 = aBase + (size_t)row0 * NN + chk0 * 8;
        const _Float16* g1 = aBase + (size_t)row1 * NN + chk1 * 8;
        unsigned l0 = ldsWave + (unsigned)(row0 * (A_ROWPAD * 2) + chk0 * 16);
        unsigned l1 = ldsWave + (unsigned)(row1 * (A_ROWPAD * 2) + chk1 * 16);
        asm volatile("global_load_async_to_lds_b128 %0, %1, off" :: "v"(l0), "v"(g0) : "memory");
        asm volatile("global_load_async_to_lds_b128 %0, %1, off" :: "v"(l1), "v"(g1) : "memory");
    }

    for (int step = 0; step < NN / 32; ++step) {
        const int m0  = step * 32;
        const int buf = step & 1;

        if (step + 1 < NN / 32) {
            // issue async copy of the NEXT A tile into the other buffer,
            // then wait with 2 outstanding -> current buffer is complete.
            const int m1 = m0 + 32;
            const _Float16* g0 = aBase + (size_t)row0 * NN + m1 + chk0 * 8;
            const _Float16* g1 = aBase + (size_t)row1 * NN + m1 + chk1 * 8;
            unsigned l0 = ldsWave + (unsigned)((buf ^ 1) * A_BUFSZ + row0 * (A_ROWPAD * 2) + chk0 * 16);
            unsigned l1 = ldsWave + (unsigned)((buf ^ 1) * A_BUFSZ + row1 * (A_ROWPAD * 2) + chk1 * 16);
            asm volatile("global_load_async_to_lds_b128 %0, %1, off" :: "v"(l0), "v"(g0) : "memory");
            asm volatile("global_load_async_to_lds_b128 %0, %1, off" :: "v"(l1), "v"(g1) : "memory");
            asm volatile("s_wait_asynccnt 0x2" ::: "memory");
        } else {
            asm volatile("s_wait_asynccnt 0x0" ::: "memory");
        }

        // A-operand from LDS (16-bit A 16x32 layout):
        //   lane<16 : row r, K = 0..7 | 16..23 ; lane>=16 : row r, K = 8..15 | 24..31
        H a;
        const _Float16* ap = &s_a[wave][buf][r][hi * 8];
        a.h[0] = *(const v8h*)(ap);
        a.h[1] = *(const v8h*)(ap + 16);

        // 8 B-operands + 8 WMMAs (B 32x16: lane<16 col t=r K=0..7 | 16..23, lane>=16 K=8..15 | 24..31)
        const _Float16* yp = yBase + m0 + hi * 8;
        #pragma unroll
        for (int oi = 0; oi < 8; ++oi) {
            H bm;
            bm.h[0] = *(const v8h*)(yp + (size_t)oi * oStride);
            bm.h[1] = *(const v8h*)(yp + (size_t)oi * oStride + 16);
            acc[oi] = __builtin_amdgcn_wmma_f32_16x16x32_f16(
                false, a.v, false, bm.v, (short)0, acc[oi], false, false);
        }
    }

    // D layout: VGPR i -> row n0 + i + hi*8, col t = r
    #pragma unroll
    for (int oi = 0; oi < 8; ++oi) {
        #pragma unroll
        for (int i = 0; i < 8; ++i) {
            out[(((size_t)b * NC + o0 + oi) * NN + n0 + hi * 8 + i) * NT + r] = acc[oi][i];
        }
    }
}

// ---------------------------------------------------------------------------
extern "C" void kernel_launch(void* const* d_in, const int* in_sizes, int n_in,
                              void* d_out, int out_size, void* d_ws, size_t ws_size,
                              hipStream_t stream) {
    const float* x     = (const float*)d_in[0];
    const float* adj   = (const float*)d_in[1];
    const float* alpha = (const float*)d_in[2];
    const float* Watt  = (const float*)d_in[3];
    const float* Wout  = (const float*)d_in[4];
    float* out = (float*)d_out;

    char* ws = (char*)d_ws;
    float*     kbuf = (float*)(ws);                          //  2 MiB
    float*     qbuf = (float*)(ws + (2ull  << 20));          //  2 MiB
    _Float16*  att  = (_Float16*)(ws + (4ull  << 20));       // 64 MiB
    _Float16*  yT   = (_Float16*)(ws + (68ull << 20));       // 64 MiB

    gacn_prep1<<<dim3(NN / 16, NB), 256, 0, stream>>>(x, alpha, Watt, kbuf, qbuf);
    gacn_prep2<<<dim3(NN / 8, NB), 256, 0, stream>>>(x, Wout, yT);
    gacn_att  <<<dim3(NN, NB), 256, 0, stream>>>(kbuf, qbuf, adj, att);
    gacn_wmma <<<dim3(NB, NN / 16, 2), 128, 0, stream>>>(att, yT, out);
}
